// SelfAttention_75960791597261
// MI455X (gfx1250) — compile-verified, matrix operand order checked by reading, not run on hardware
//
#include <hip/hip_runtime.h>
#include <hip/hip_bf16.h>

// ---------------------------------------------------------------------------
// Snake-attention forward for MI455X (gfx1250, wave32, WMMA).
// All dense GEMMs use v_wmma_f32_16x16x32_f16 (f16 in, f32 accum).
// Irregular snake parts (3x3 convs on 8x8 maps, groupnorm, tanh, deformable
// bilinear sampling) run as per-(b,h) workgroup VALU kernels with LDS staging.
// ---------------------------------------------------------------------------

#define BQ 128
#define SEQ 64
#define DM 512
#define NH 8
#define BH (BQ * NH)          // 1024 heads
#define NCH 64                // snake channel count N = VA*HA

typedef __attribute__((ext_vector_type(16))) _Float16 v16h;
typedef __attribute__((ext_vector_type(8)))  float    v8f;

union Frag16 { v16h v; unsigned u[8]; _Float16 h[16]; };

// A/B fragment, contiguous-K layout (row-major [rows][ld] f16):
// lane L holds row (row0 + L%16); per the CDNA5 16-bit 16x32 layout, dword v
// (v=0..3) holds K = 2v + 8*(L/16), dwords 4..7 hold K+16.
__device__ __forceinline__ v16h ld_frag_rowK(const _Float16* __restrict__ base,
                                             int row0, int ld, int k0, int lane) {
  Frag16 f;
  const int hl = lane >> 4;
  const _Float16* p = base + (long)(row0 + (lane & 15)) * ld + k0 + hl * 8;
#pragma unroll
  for (int v = 0; v < 4; ++v) {
    f.u[v]     = *(const unsigned*)(p + 2 * v);
    f.u[v + 4] = *(const unsigned*)(p + 16 + 2 * v);
  }
  return f.v;
}

// B fragment gathered from a row-major [K][ld] matrix (element (n,k) = base[k*ld+n]).
__device__ __forceinline__ v16h ld_frag_colK_h(const _Float16* __restrict__ base,
                                               int col0, int ld, int k0, int lane) {
  Frag16 f;
  const int hl = lane >> 4;
  const int col = col0 + (lane & 15);
#pragma unroll
  for (int e = 0; e < 16; ++e) {
    int v = e >> 1;
    int k = ((v & 3) * 2) + ((v >> 2) * 16) + hl * 8 + (e & 1);
    f.h[e] = base[(long)(k0 + k) * ld + col];
  }
  return f.v;
}

__device__ __forceinline__ v16h ld_frag_colK_f(const float* __restrict__ base,
                                               int col0, int ld, int k0, int lane) {
  Frag16 f;
  const int hl = lane >> 4;
  const int col = col0 + (lane & 15);
#pragma unroll
  for (int e = 0; e < 16; ++e) {
    int v = e >> 1;
    int k = ((v & 3) * 2) + ((v >> 2) * 16) + hl * 8 + (e & 1);
    f.h[e] = (_Float16)base[(long)(k0 + k) * ld + col];
  }
  return f.v;
}

// ---------------------------------------------------------------------------
__global__ void k_cast_f16(const float* __restrict__ in, _Float16* __restrict__ out, int n) {
  int i = blockIdx.x * blockDim.x + threadIdx.x;
  if (i < n) out[i] = (_Float16)in[i];
}

// out[c][r] = (f16) in[r][c]   (weight transpose so B fragments load contiguously)
__global__ void k_transpose_f16(const float* __restrict__ in, _Float16* __restrict__ out,
                                int rows, int cols) {
  int i = blockIdx.x * blockDim.x + threadIdx.x;
  if (i < rows * cols) {
    int r = i / cols, c = i % cols;
    out[(long)c * rows + r] = (_Float16)in[i];
  }
}

// ---------------------------------------------------------------------------
// Generic WMMA GEMM: C = (A[MxK] * Bt[NxK]^T + bias) * scale
// grid = (N/64, M/16), block = 128 (4 waves, 1 tile each).
// layout 0: row-major [M][Ncols]
// layout 1: qkv  -> out[((b*8+h)*64+s)*64+dk]   (b=m/64,s=m%64,h=n/64,dk=n%64)
// layout 2: vT   -> out[((b*8+h)*64+dk)*64+s]
__global__ void k_gemm_wmma(const _Float16* __restrict__ A, const _Float16* __restrict__ Bt,
                            const float* __restrict__ bias, float scale, int Kdim, int Ncols,
                            int layout, float* __restrict__ outF, _Float16* __restrict__ outH) {
  const int lane = threadIdx.x & 31;
  const int wave = threadIdx.x >> 5;
  const int m0 = blockIdx.y * 16;
  const int n0 = blockIdx.x * 64 + wave * 16;
  v8f acc = {};
  for (int k0 = 0; k0 < Kdim; k0 += 32) {
    v16h a = ld_frag_rowK(A, m0, Kdim, k0, lane);
    v16h b = ld_frag_rowK(Bt, n0, Kdim, k0, lane);
    acc = __builtin_amdgcn_wmma_f32_16x16x32_f16(false, a, false, b, (short)0, acc, false, false);
  }
  const int hl = lane >> 4;
  const int nn = n0 + (lane & 15);
  const float bn = bias ? bias[nn] : 0.0f;
#pragma unroll
  for (int r = 0; r < 8; ++r) {
    int m = m0 + r + 8 * hl;
    float v = (acc[r] + bn) * scale;
    long off;
    if (layout == 0) {
      off = (long)m * Ncols + nn;
    } else {
      int b = m >> 6, s = m & 63, h = nn >> 6, dk = nn & 63;
      off = (layout == 1) ? (((long)(b * 8 + h) * 64 + s) * 64 + dk)
                          : (((long)(b * 8 + h) * 64 + dk) * 64 + s);
    }
    if (outF) outF[off] = v;
    if (outH) outH[off] = (_Float16)v;
  }
}

// att = q*k^T per head (scale already folded into q).  1 block / head, 8 waves x 2 tiles.
__global__ void k_attn_qk(const _Float16* __restrict__ qh, const _Float16* __restrict__ kh,
                          float* __restrict__ att, _Float16* __restrict__ atth) {
  const int bh = blockIdx.x;
  const int lane = threadIdx.x & 31;
  const int wave = threadIdx.x >> 5;
  const _Float16* qb = qh + (long)bh * 4096;
  const _Float16* kb = kh + (long)bh * 4096;
  float* ab = att + (long)bh * 4096;
  _Float16* ah = atth + (long)bh * 4096;
  for (int t = wave; t < 16; t += 8) {
    int m0 = (t >> 2) * 16, n0 = (t & 3) * 16;
    v8f acc = {};
#pragma unroll
    for (int k0 = 0; k0 < 64; k0 += 32) {
      v16h a = ld_frag_rowK(qb, m0, 64, k0, lane);
      v16h b = ld_frag_rowK(kb, n0, 64, k0, lane);
      acc = __builtin_amdgcn_wmma_f32_16x16x32_f16(false, a, false, b, (short)0, acc, false, false);
    }
    int hl = lane >> 4, nn = n0 + (lane & 15);
#pragma unroll
    for (int r = 0; r < 8; ++r) {
      int m = m0 + r + 8 * hl;
      ab[m * 64 + nn] = acc[r];
      ah[m * 64 + nn] = (_Float16)acc[r];
    }
  }
}

// x0[n][c] = sum_q p1_w[q][n]*att[q][c] + p1_b[n]   (A = p1^T, B gathered from att^T)
__global__ void k_snake_p1(const _Float16* __restrict__ p1T, const _Float16* __restrict__ atth,
                           const float* __restrict__ p1b, float* __restrict__ x0) {
  const int bh = blockIdx.x;
  const int lane = threadIdx.x & 31;
  const int wave = threadIdx.x >> 5;
  const _Float16* ab = atth + (long)bh * 4096;
  float* xb = x0 + (long)bh * 4096;
  for (int t = wave; t < 16; t += 8) {
    int m0 = (t >> 2) * 16, n0 = (t & 3) * 16;
    v8f acc = {};
#pragma unroll
    for (int k0 = 0; k0 < 64; k0 += 32) {
      v16h a = ld_frag_rowK(p1T, m0, 64, k0, lane);
      v16h b = ld_frag_colK_h(ab, n0, 64, k0, lane);
      acc = __builtin_amdgcn_wmma_f32_16x16x32_f16(false, a, false, b, (short)0, acc, false, false);
    }
    int hl = lane >> 4, nn = n0 + (lane & 15);
#pragma unroll
    for (int r = 0; r < 8; ++r) {
      int m = m0 + r + 8 * hl;
      xb[m * 64 + nn] = acc[r] + p1b[m];
    }
  }
}

// att[m][c] += sum_q p2_w[q][m]*xo[q][c] + p2_b[m]
__global__ void k_snake_p2_add(const _Float16* __restrict__ p2T, const float* __restrict__ xo,
                               const float* __restrict__ p2b, float* __restrict__ att) {
  const int bh = blockIdx.x;
  const int lane = threadIdx.x & 31;
  const int wave = threadIdx.x >> 5;
  const float* xb = xo + (long)bh * 4096;
  float* ab = att + (long)bh * 4096;
  for (int t = wave; t < 16; t += 8) {
    int m0 = (t >> 2) * 16, n0 = (t & 3) * 16;
    v8f acc = {};
#pragma unroll
    for (int k0 = 0; k0 < 64; k0 += 32) {
      v16h a = ld_frag_rowK(p2T, m0, 64, k0, lane);
      v16h b = ld_frag_colK_f(xb, n0, 64, k0, lane);
      acc = __builtin_amdgcn_wmma_f32_16x16x32_f16(false, a, false, b, (short)0, acc, false, false);
    }
    int hl = lane >> 4, nn = n0 + (lane & 15);
#pragma unroll
    for (int r = 0; r < 8; ++r) {
      int m = m0 + r + 8 * hl;
      ab[m * 64 + nn] += acc[r] + p2b[m];
    }
  }
}

// ctx = softmax(att) * v ; output f16 at [b*64+s][512] col h*64+dk
__global__ void k_attn_v(const _Float16* __restrict__ atth, const _Float16* __restrict__ vTh,
                         _Float16* __restrict__ ctxh) {
  const int bh = blockIdx.x;
  const int b = bh >> 3, h = bh & 7;
  const int lane = threadIdx.x & 31;
  const int wave = threadIdx.x >> 5;
  const _Float16* ab = atth + (long)bh * 4096;
  const _Float16* vb = vTh + (long)bh * 4096;
  for (int t = wave; t < 16; t += 8) {
    int m0 = (t >> 2) * 16, n0 = (t & 3) * 16;
    v8f acc = {};
#pragma unroll
    for (int k0 = 0; k0 < 64; k0 += 32) {
      v16h a = ld_frag_rowK(ab, m0, 64, k0, lane);
      v16h bfrag = ld_frag_rowK(vb, n0, 64, k0, lane);
      acc = __builtin_amdgcn_wmma_f32_16x16x32_f16(false, a, false, bfrag, (short)0, acc, false, false);
    }
    int hl = lane >> 4, nn = n0 + (lane & 15);
#pragma unroll
    for (int r = 0; r < 8; ++r) {
      int m = m0 + r + 8 * hl;
      ctxh[(long)(b * 64 + m) * 512 + h * 64 + nn] = (_Float16)acc[r];
    }
  }
}

// ---------------------------------------------------------------------------
// 3x3 conv (pad 1) on 8x8 maps + group-norm + activation. 1 block / head.
// in channels from up to 3 concatenated [1024][64][64] buffers. act: 0=relu 1=tanh
__global__ void k_conv3x3_gn(const float* __restrict__ in0, const float* __restrict__ in1,
                             const float* __restrict__ in2, int Cin,
                             const float* __restrict__ w, const float* __restrict__ bias,
                             const float* __restrict__ gamma, const float* __restrict__ beta,
                             int Cout, int groups, int act, float* __restrict__ out) {
  const int bh = blockIdx.x;
  __shared__ float s_out[4096];
  __shared__ float s_mu[16], s_rs[16];
  const int tid = threadIdx.x;
  for (int idx = tid; idx < Cout * 64; idx += 256) {
    int co = idx >> 6, p = idx & 63, py = p >> 3, px = p & 7;
    float acc = bias[co];
    for (int ci = 0; ci < Cin; ++ci) {
      const float* src = (ci < 64 ? in0 : (ci < 128 ? in1 : in2)) + (long)bh * 4096 + (ci & 63) * 64;
      const float* wr = w + ((long)co * Cin + ci) * 9;
#pragma unroll
      for (int dy = -1; dy <= 1; ++dy) {
        int yy = py + dy; if ((unsigned)yy >= 8u) continue;
#pragma unroll
        for (int dx = -1; dx <= 1; ++dx) {
          int xx = px + dx; if ((unsigned)xx >= 8u) continue;
          acc += src[yy * 8 + xx] * wr[(dy + 1) * 3 + (dx + 1)];
        }
      }
    }
    s_out[idx] = acc;
  }
  __syncthreads();
  const int gs = Cout / groups;
  if (tid < groups) {
    float s = 0.f, s2 = 0.f;
    const int cnt = gs * 64;
    const int base = tid * cnt;
    for (int i = 0; i < cnt; ++i) { float v = s_out[base + i]; s += v; s2 += v * v; }
    float mu = s / cnt, var = s2 / cnt - mu * mu;
    s_mu[tid] = mu; s_rs[tid] = rsqrtf(var + 1e-5f);
  }
  __syncthreads();
  for (int idx = tid; idx < Cout * 64; idx += 256) {
    int co = idx >> 6, g = co / gs;
    float v = (s_out[idx] - s_mu[g]) * s_rs[g] * gamma[co] + beta[co];
    v = (act == 0) ? fmaxf(v, 0.f) : tanhf(v);
    out[(long)bh * Cout * 64 + idx] = v;
  }
}

// Deformable snake conv: cumulative offsets -> bilinear sample -> Kx1/1xK conv
// -> group-norm(16) -> relu.  1 block (256 thr) / head; 4 threads per pixel.
__global__ void k_dsconv(const float* __restrict__ x0, const float* __restrict__ off,
                         const float* __restrict__ w, const float* __restrict__ bias,
                         const float* __restrict__ gamma, const float* __restrict__ beta,
                         int morph, float* __restrict__ out) {
  const int bh = blockIdx.x;
  __shared__ float s_x[4096];
  __shared__ float s_cum[9 * 64];
  __shared__ float s_w[4096];
  __shared__ float s_out[4096];
  __shared__ float s_mu[16], s_rs[16];
  const int tid = threadIdx.x;
  const float* xb = x0 + (long)bh * 4096;
  for (int i = tid; i < 4096; i += 256) s_x[i] = xb[i];
  if (tid < 64) {
    const float* ob = off + (long)bh * 18 * 64 + (morph ? 9 * 64 : 0);
    float c = 0.f;
    s_cum[4 * 64 + tid] = 0.f;
    for (int kk = 5; kk <= 8; ++kk) { c += ob[kk * 64 + tid]; s_cum[kk * 64 + tid] = c; }
    c = 0.f;
    for (int kk = 3; kk >= 0; --kk) { c += ob[kk * 64 + tid]; s_cum[kk * 64 + tid] = c; }
  }
  __syncthreads();
  const int p = tid >> 2;        // pixel 0..63
  const int qtr = tid & 3;       // this thread handles c_out qtr*16..qtr*16+15
  const int py = p >> 3, px = p & 7;
  float acc[16];
#pragma unroll
  for (int j = 0; j < 16; ++j) acc[j] = bias[qtr * 16 + j];
  for (int kk = 0; kk < 9; ++kk) {
    for (int i = tid; i < 4096; i += 256) s_w[i] = w[(long)i * 9 + kk];
    __syncthreads();
    float fy, fx;
    if (morph == 0) { fy = (float)py + s_cum[kk * 64 + p]; fx = (float)(px + kk - 4); }
    else            { fy = (float)(py + kk - 4); fx = (float)px + s_cum[kk * 64 + p]; }
    fy = fminf(fmaxf(fy, 0.f), 7.f);
    fx = fminf(fmaxf(fx, 0.f), 7.f);
    float y0f = floorf(fy), x0f = floorf(fx);
    float wy = fy - y0f, wx = fx - x0f;
    int y0i = (int)y0f, x0i = (int)x0f;
    int y1i = min(y0i + 1, 7), x1i = min(x0i + 1, 7);
    int i00 = y0i * 8 + x0i, i01 = y0i * 8 + x1i, i10 = y1i * 8 + x0i, i11 = y1i * 8 + x1i;
    float w00 = (1.f - wy) * (1.f - wx), w01 = (1.f - wy) * wx;
    float w10 = wy * (1.f - wx), w11 = wy * wx;
    for (int ci = 0; ci < 64; ++ci) {
      const float* row = s_x + ci * 64;
      float f = row[i00] * w00 + row[i01] * w01 + row[i10] * w10 + row[i11] * w11;
      const float* ws = s_w + (qtr * 16) * 64 + ci;
#pragma unroll
      for (int j = 0; j < 16; ++j) acc[j] += ws[j * 64] * f;
    }
    __syncthreads();
  }
#pragma unroll
  for (int j = 0; j < 16; ++j) s_out[(qtr * 16 + j) * 64 + p] = acc[j];
  __syncthreads();
  if (tid < 16) {
    float s = 0.f, s2 = 0.f;
    const int base = tid * 256;
    for (int i = 0; i < 256; ++i) { float v = s_out[base + i]; s += v; s2 += v * v; }
    float mu = s / 256.f, var = s2 / 256.f - mu * mu;
    s_mu[tid] = mu; s_rs[tid] = rsqrtf(var + 1e-5f);
  }
  __syncthreads();
  for (int idx = tid; idx < 4096; idx += 256) {
    int co = idx >> 6, g = co >> 2;
    float v = (s_out[idx] - s_mu[g]) * s_rs[g] * gamma[co] + beta[co];
    out[(long)bh * 4096 + idx] = fmaxf(v, 0.f);
  }
}

// Row softmax over 64 entries; 64 threads / head, one row each.
__global__ void k_softmax(float* __restrict__ att, _Float16* __restrict__ atth) {
  const int bh = blockIdx.x, q = threadIdx.x;
  float* row = att + (long)bh * 4096 + q * 64;
  _Float16* hr = atth + (long)bh * 4096 + q * 64;
  float mx = -1e30f;
#pragma unroll
  for (int i = 0; i < 64; ++i) mx = fmaxf(mx, row[i]);
  float e[64]; float s = 0.f;
#pragma unroll
  for (int i = 0; i < 64; ++i) { e[i] = expf(row[i] - mx); s += e[i]; }
  float inv = 1.f / s;
#pragma unroll
  for (int i = 0; i < 64; ++i) { float v = e[i] * inv; row[i] = v; hr[i] = (_Float16)v; }
}

// ---------------------------------------------------------------------------
extern "C" void kernel_launch(void* const* d_in, const int* in_sizes, int n_in,
                              void* d_out, int out_size, void* d_ws, size_t ws_size,
                              hipStream_t stream) {
  // Inputs flattened in setup_inputs() dict insertion order (nested dicts depth-first):
  const float* x     = (const float*)d_in[0];
  const float* Wq    = (const float*)d_in[1];  const float* bq = (const float*)d_in[2];
  const float* Wk    = (const float*)d_in[3];  const float* bk = (const float*)d_in[4];
  const float* Wv    = (const float*)d_in[5];  const float* bv = (const float*)d_in[6];
  const float* Wo    = (const float*)d_in[7];  const float* bo = (const float*)d_in[8];
  const float* p1w   = (const float*)d_in[9];  const float* p1b = (const float*)d_in[10];
  const float* p2w   = (const float*)d_in[11]; const float* p2b = (const float*)d_in[12];
  const float* c0w   = (const float*)d_in[13]; const float* c0b = (const float*)d_in[14];
  const float* c0g   = (const float*)d_in[15]; const float* c0be = (const float*)d_in[16];
  const float* c1w   = (const float*)d_in[17]; const float* c1b = (const float*)d_in[18];
  const float* c1g   = (const float*)d_in[19]; const float* c1be = (const float*)d_in[20];
  const float* dxow  = (const float*)d_in[21]; const float* dxob = (const float*)d_in[22];
  const float* dxgg  = (const float*)d_in[23]; const float* dxgb = (const float*)d_in[24];
  const float* dxw   = (const float*)d_in[25]; const float* dxb = (const float*)d_in[26];
  const float* dxng  = (const float*)d_in[27]; const float* dxnb = (const float*)d_in[28];
  const float* dyow  = (const float*)d_in[29]; const float* dyob = (const float*)d_in[30];
  const float* dygg  = (const float*)d_in[31]; const float* dygb = (const float*)d_in[32];
  const float* dyw   = (const float*)d_in[33]; const float* dyb = (const float*)d_in[34];
  const float* dyng  = (const float*)d_in[35]; const float* dynb = (const float*)d_in[36];

  // Workspace carve-out (~141 MB).
  char* ws = (char*)d_ws;
  size_t cur = 0;
  auto carve = [&](size_t bytes) -> void* {
    void* p = ws + cur;
    cur = (cur + bytes + 255) & ~(size_t)255;
    return p;
  };
  _Float16* xh   = (_Float16*)carve((size_t)8192 * 512 * 2);
  _Float16* WqT  = (_Float16*)carve((size_t)512 * 512 * 2);
  _Float16* WkT  = (_Float16*)carve((size_t)512 * 512 * 2);
  _Float16* WvT  = (_Float16*)carve((size_t)512 * 512 * 2);
  _Float16* WoT  = (_Float16*)carve((size_t)512 * 512 * 2);
  _Float16* p1T  = (_Float16*)carve((size_t)64 * 64 * 2);
  _Float16* p2T  = (_Float16*)carve((size_t)64 * 64 * 2);
  _Float16* qh   = (_Float16*)carve((size_t)BH * 4096 * 2);
  _Float16* kh   = (_Float16*)carve((size_t)BH * 4096 * 2);
  _Float16* vTh  = (_Float16*)carve((size_t)BH * 4096 * 2);
  float*    att  = (float*)   carve((size_t)BH * 4096 * 4);
  _Float16* atth = (_Float16*)carve((size_t)BH * 4096 * 2);
  float*    x0   = (float*)   carve((size_t)BH * 4096 * 4);   // also reused as xo
  float*    offb = (float*)   carve((size_t)BH * 18 * 64 * 4);
  float*    x1   = (float*)   carve((size_t)BH * 4096 * 4);
  float*    x2   = (float*)   carve((size_t)BH * 4096 * 4);
  float*    x3   = (float*)   carve((size_t)BH * 4096 * 4);
  _Float16* ctxh = (_Float16*)carve((size_t)8192 * 512 * 2);
  (void)ws_size; (void)n_in; (void)in_sizes; (void)out_size;

  const float inv_sqrt_dk = 0.125f;   // 1/sqrt(64), folded into q

  // 1) activation + weight precision prep
  k_cast_f16<<<(8192 * 512 + 255) / 256, 256, 0, stream>>>(x, xh, 8192 * 512);
  k_transpose_f16<<<1024, 256, 0, stream>>>(Wq, WqT, 512, 512);
  k_transpose_f16<<<1024, 256, 0, stream>>>(Wk, WkT, 512, 512);
  k_transpose_f16<<<1024, 256, 0, stream>>>(Wv, WvT, 512, 512);
  k_transpose_f16<<<1024, 256, 0, stream>>>(Wo, WoT, 512, 512);
  k_transpose_f16<<<16, 256, 0, stream>>>(p1w, p1T, 64, 64);
  k_transpose_f16<<<16, 256, 0, stream>>>(p2w, p2T, 64, 64);

  // 2) q/k/v projections (WMMA), q pre-scaled, v stored transposed
  dim3 ggrid(512 / 64, 8192 / 16);
  k_gemm_wmma<<<ggrid, 128, 0, stream>>>(xh, WqT, bq, inv_sqrt_dk, 512, 512, 1, nullptr, qh);
  k_gemm_wmma<<<ggrid, 128, 0, stream>>>(xh, WkT, bk, 1.0f, 512, 512, 1, nullptr, kh);
  k_gemm_wmma<<<ggrid, 128, 0, stream>>>(xh, WvT, bv, 1.0f, 512, 512, 2, nullptr, vTh);

  // 3) att = q k^T (WMMA per head)
  k_attn_qk<<<BH, 256, 0, stream>>>(qh, kh, att, atth);

  // 4) snake: p1 projection (WMMA), then convs/deformable sampling
  k_snake_p1<<<BH, 256, 0, stream>>>(p1T, atth, p1b, x0);
  k_conv3x3_gn<<<BH, 256, 0, stream>>>(x0, nullptr, nullptr, 64, c0w, c0b, c0g, c0be,
                                       64, 16, 0, x1);
  k_conv3x3_gn<<<BH, 256, 0, stream>>>(x0, nullptr, nullptr, 64, dxow, dxob, dxgg, dxgb,
                                       18, 9, 1, offb);
  k_dsconv<<<BH, 256, 0, stream>>>(x0, offb, dxw, dxb, dxng, dxnb, 0, x2);
  k_conv3x3_gn<<<BH, 256, 0, stream>>>(x0, nullptr, nullptr, 64, dyow, dyob, dygg, dygb,
                                       18, 9, 1, offb);
  k_dsconv<<<BH, 256, 0, stream>>>(x0, offb, dyw, dyb, dyng, dynb, 1, x3);
  k_conv3x3_gn<<<BH, 256, 0, stream>>>(x1, x2, x3, 192, c1w, c1b, c1g, c1be,
                                       64, 16, 0, x0);   // xo -> x0

  // 5) p2 projection added into att (WMMA), softmax, att*v (WMMA)
  k_snake_p2_add<<<BH, 256, 0, stream>>>(p2T, x0, p2b, att);
  k_softmax<<<BH, 64, 0, stream>>>(att, atth);
  k_attn_v<<<BH, 256, 0, stream>>>(atth, vTh, ctxh);

  // 6) output projection (WMMA, f32 out)
  k_gemm_wmma<<<ggrid, 128, 0, stream>>>(ctxh, WoT, bo, 1.0f, 512, 512, 0,
                                         (float*)d_out, nullptr);
}
